// DeformableConv2D_11596411699679
// MI455X (gfx1250) — matrix operand order repaired and checked by reading, not compiled, not run on hardware
//
#include <hip/hip_runtime.h>
#include <math.h>

#define B_   4
#define H_   64
#define W_   64
#define C_   256
#define F_   256
#define Ksz  3
#define KK   9
#define PAD  1
#define HW   (H_ * W_)
#define NOFF 27
#define TILE_P 32
#define KCH  64                      // K-chunk per staging pass (2 WMMA K-steps)
#define NIT  (KK * (C_ / KCH))       // 36 pipeline iterations

typedef __attribute__((ext_vector_type(16))) __bf16       v16bf;
typedef __attribute__((ext_vector_type(8)))  float        v8f;
typedef __attribute__((ext_vector_type(8)))  unsigned int v8u;

__device__ __forceinline__ unsigned short f2bf(float f) {
    // round-to-nearest-even f32 -> bf16 (pure bit manipulation)
    unsigned int u = __builtin_bit_cast(unsigned int, f);
    u += 0x7FFFu + ((u >> 16) & 1u);
    return (unsigned short)(u >> 16);
}

// ---------------------------------------------------------------------------
// K1: offset/mask predictor conv (3x3, C=256 -> 27), SAME padding, NHWC.
// wi layout: [B*HW, 27]  (channel = {o1:0..8, o2:9..17, mask-logit:18..26})
// ---------------------------------------------------------------------------
__global__ void dcn_offset_conv(const float* __restrict__ x,
                                const float* __restrict__ w_offset,
                                const float* __restrict__ b_offset,
                                float* __restrict__ wi) {
    int tid = blockIdx.x * blockDim.x + threadIdx.x;
    if (tid >= B_ * HW * NOFF) return;
    int oc  = tid % NOFF;
    int pix = tid / NOFF;
    int b   = pix / HW;
    int hw  = pix % HW;
    int h   = hw / W_, w = hw % W_;
    float sum = b_offset[oc];
    for (int ky = 0; ky < Ksz; ++ky) {
        int y = h - PAD + ky;
        if (y < 0 || y >= H_) continue;
        for (int kx = 0; kx < Ksz; ++kx) {
            int xq = w - PAD + kx;
            if (xq < 0 || xq >= W_) continue;
            const float* xp = x + (size_t)(b * HW + y * W_ + xq) * C_;
            const float* wp = w_offset + (size_t)((ky * Ksz + kx) * C_) * NOFF + oc;
#pragma unroll 4
            for (int c = 0; c < C_; ++c) sum += xp[c] * wp[c * NOFF];
        }
    }
    wi[tid] = sum;
}

// ---------------------------------------------------------------------------
// K2: filter reorder + bf16 quantize: filt[F,C,KK] f32 -> filt_r[KK,F,C] bf16
// ---------------------------------------------------------------------------
__global__ void dcn_filt_reorder(const float* __restrict__ filt,
                                 unsigned short* __restrict__ filt_r) {
    int idx = blockIdx.x * blockDim.x + threadIdx.x;
    if (idx >= F_ * C_ * KK) return;
    int f  = idx / (C_ * KK);
    int r  = idx % (C_ * KK);
    int c  = r / KK;
    int kk = r % KK;
    filt_r[((size_t)kk * F_ + f) * C_ + c] = f2bf(filt[idx]);
}

// ---------------------------------------------------------------------------
// Staging for pipeline iteration `it` into LDS buffer `buf`:
//   B tile: filt_r[kk][f=tid][c0..c0+63]  (128B contiguous per thread)
//   A tile: bilinear-sample 8 channels of one (pixel, tap) per thread
// ---------------------------------------------------------------------------
__device__ __forceinline__ void stage_tiles(int it, int tid,
                                            unsigned short* __restrict__ s_a,
                                            unsigned short* __restrict__ s_b,
                                            const float* __restrict__ x,
                                            const unsigned short* __restrict__ filt_r,
                                            const float* __restrict__ s_w,
                                            const int* __restrict__ s_idx) {
    const int kk = it >> 2;          // tap
    const int c0 = (it & 3) << 6;    // channel chunk base
    // ---- B tile ----
    {
        const unsigned short* srcp = filt_r + ((size_t)kk * F_ + tid) * C_ + c0;
        const uint4* src = (const uint4*)srcp;
        uint4*       dst = (uint4*)(s_b + tid * KCH);
#pragma unroll
        for (int q = 0; q < 8; ++q) dst[q] = src[q];
        __builtin_prefetch(srcp + KCH, 0, 1);            // next chunk -> global_prefetch_b8
    }
    // ---- A tile ----
    {
        const int pa    = tid >> 3;          // pixel 0..31
        const int ca    = (tid & 7) * 8;     // channel 0,8,...,56
        const int entry = pa * KK + kk;
        float4 v0 = make_float4(0.f, 0.f, 0.f, 0.f);
        float4 v1 = v0;
#pragma unroll
        for (int j = 0; j < 4; ++j) {
            float wgt = s_w[j * (TILE_P * KK) + entry];
            const float* base = x + s_idx[j * (TILE_P * KK) + entry] + c0 + ca;
            const float4 a = *(const float4*)base;
            const float4 c = *(const float4*)(base + 4);
            v0.x += wgt * a.x; v0.y += wgt * a.y; v0.z += wgt * a.z; v0.w += wgt * a.w;
            v1.x += wgt * c.x; v1.y += wgt * c.y; v1.z += wgt * c.z; v1.w += wgt * c.w;
        }
        unsigned short* ap = s_a + pa * KCH + ca;
        ap[0] = f2bf(v0.x); ap[1] = f2bf(v0.y); ap[2] = f2bf(v0.z); ap[3] = f2bf(v0.w);
        ap[4] = f2bf(v1.x); ap[5] = f2bf(v1.y); ap[6] = f2bf(v1.z); ap[7] = f2bf(v1.w);
    }
}

// ---------------------------------------------------------------------------
// K3: fused sampling + im2col GEMM, double-buffered LDS, bf16 WMMA.
// 512 WGs x 256 threads (8 waves). WG = 32 pixels x all 256 F.
// Per iteration: 8 wmma per wave, ONE workgroup barrier (software pipeline).
// ---------------------------------------------------------------------------
__launch_bounds__(256, 2)
__global__ void dcn_wmma_main(const float* __restrict__ x,
                              const float* __restrict__ wi,
                              const unsigned short* __restrict__ filt_r,
                              float* __restrict__ out) {
    __shared__ float          s_w[4 * TILE_P * KK];
    __shared__ int            s_idx[4 * TILE_P * KK];
    __shared__ __align__(32) unsigned short s_a[2][TILE_P * KCH];  // 2 x 4KB
    __shared__ __align__(32) unsigned short s_b[2][F_ * KCH];      // 2 x 32KB

    const int tid  = threadIdx.x;
    const int pix0 = blockIdx.x * TILE_P;    // HW % 32 == 0 -> one batch per WG
    const int b    = pix0 / HW;

    // ---- Phase 1: per (pixel, tap) sampling parameters ----
    for (int idx = tid; idx < TILE_P * KK; idx += 256) {
        int pl = idx / KK, tap = idx % KK;
        int pix = pix0 + pl;
        int hw  = pix % HW;
        int h   = hw / W_, w = hw % W_;
        const float* wip = wi + (size_t)pix * NOFF;
        float o1 = wip[tap];
        float o2 = wip[KK + tap];
        float m  = 1.0f / (1.0f + __expf(-wip[2 * KK + tap]));   // sigmoid mask
        int ki = tap / Ksz, kj = tap % Ksz;
        float py  = (float)(h - PAD + ki) + o1;
        float px  = (float)(w - PAD + kj) + o2;
        float y0f = floorf(py), x0f = floorf(px);
        float dy  = py - y0f,   dx  = px - x0f;
        int   y0  = (int)y0f,   x0  = (int)x0f;
#pragma unroll
        for (int j = 0; j < 4; ++j) {
            int   yy = y0 + (j >> 1);
            int   xx = x0 + (j & 1);
            float wy = (j >> 1) ? dy : (1.0f - dy);
            float wx = (j & 1) ? dx : (1.0f - dx);
            bool  valid = (yy >= 0) && (yy < H_) && (xx >= 0) && (xx < W_);
            int yc = yy < 0 ? 0 : (yy > H_ - 1 ? H_ - 1 : yy);
            int xc = xx < 0 ? 0 : (xx > W_ - 1 ? W_ - 1 : xx);
            s_w[j * (TILE_P * KK) + idx]   = valid ? (wy * wx * m) : 0.0f;
            s_idx[j * (TILE_P * KK) + idx] = (b * HW + yc * W_ + xc) * C_;
        }
    }
    __syncthreads();

    const int lane   = tid & 31;
    const int wave   = tid >> 5;
    const int wave_m = wave >> 2;   // 0..1 : 16-pixel m-tile
    const int wave_n = wave & 3;    // 0..3 : 64-F  n-range
    const int hi     = lane >> 4;   // lane half (K-split per WMMA layout)
    const int ml     = lane & 15;

    v8f zero = {};
    v8f acc[4];
#pragma unroll
    for (int t = 0; t < 4; ++t) acc[t] = zero;

    // ---- prologue: stage iteration 0 into buffer 0 ----
    stage_tiles(0, tid, s_a[0], s_b[0], x, filt_r, s_w, s_idx);
    __syncthreads();

    const int arow = wave_m * 16 + ml;

    for (int it = 0; it < NIT; ++it) {
        const int cur = it & 1;
        // stage next iteration into the other buffer (overlaps with WMMA below)
        if (it + 1 < NIT)
            stage_tiles(it + 1, tid, s_a[1 - cur], s_b[1 - cur], x, filt_r, s_w, s_idx);

        const unsigned short* sa = s_a[cur];
        const unsigned short* sb = s_b[cur];
#pragma unroll
        for (int h2 = 0; h2 < 2; ++h2) {           // two K=32 steps per chunk
            // A fragment (16x32 bf16: lanes<16 K={0..15}, lanes>=16 K={16..31})
            const uint4 alo = *(const uint4*)(sa + arow * KCH + h2 * 32 + hi * 16);
            const uint4 ahi = *(const uint4*)(sa + arow * KCH + h2 * 32 + hi * 16 + 8);
            v8u au = { alo.x, alo.y, alo.z, alo.w, ahi.x, ahi.y, ahi.z, ahi.w };
            v16bf afrag = __builtin_bit_cast(v16bf, au);
#pragma unroll
            for (int t = 0; t < 4; ++t) {
                int n = wave_n * 64 + t * 16 + ml;
                const v8u bu = *(const v8u*)(sb + n * KCH + h2 * 32 + hi * 16);
                v16bf bfrag = __builtin_bit_cast(v16bf, bu);
                acc[t] = __builtin_amdgcn_wmma_f32_16x16x32_bf16(
                    false, afrag, false, bfrag, (short)0, acc[t], false, false);
            }
        }
        __syncthreads();   // single barrier per iteration (double-buffered)
    }

    // ---- epilogue: C/D layout (VGPR v: lanes<16 M=v, lanes>=16 M=v+8) -> NHWC ----
#pragma unroll
    for (int t = 0; t < 4; ++t) {
#pragma unroll
        for (int v = 0; v < 8; ++v) {
            int m_local = v + 8 * hi;
            int n       = wave_n * 64 + t * 16 + ml;
            int pix     = pix0 + wave_m * 16 + m_local;
            out[(size_t)pix * F_ + n] = acc[t][v];
        }
    }
}

// ---------------------------------------------------------------------------
extern "C" void kernel_launch(void* const* d_in, const int* in_sizes, int n_in,
                              void* d_out, int out_size, void* d_ws, size_t ws_size,
                              hipStream_t stream) {
    const float* x        = (const float*)d_in[0];   // [4,64,64,256]
    const float* w_offset = (const float*)d_in[1];   // [3,3,256,27]
    const float* b_offset = (const float*)d_in[2];   // [27]
    const float* filt     = (const float*)d_in[3];   // [256,256,3,3]
    float* out = (float*)d_out;                      // [4,64,64,256]

    const size_t WI_BYTES = (size_t)B_ * HW * NOFF * sizeof(float);
    float*          wi     = (float*)d_ws;
    unsigned short* filt_r = (unsigned short*)((char*)d_ws + WI_BYTES);

    {
        int total = B_ * HW * NOFF;
        dcn_offset_conv<<<(total + 255) / 256, 256, 0, stream>>>(x, w_offset, b_offset, wi);
    }
    {
        int total = F_ * C_ * KK;
        dcn_filt_reorder<<<(total + 255) / 256, 256, 0, stream>>>(filt, filt_r);
    }
    {
        dcn_wmma_main<<<(B_ * HW) / TILE_P, 256, 0, stream>>>(x, wi, filt_r, out);
    }
}
